// T3KAN_80479097193101
// MI455X (gfx1250) — matrix operand order repaired and checked by reading, not compiled, not run on hardware
//
#include <hip/hip_runtime.h>
#include <hip/hip_bf16.h>
#include <stdint.h>

#define Bz   4
#define Sz   2048
#define DIMz 1024
#define Hz   16
#define HDz  64
#define MBz  16
#define NMBz 128
#define NBz  11
#define NKz  15
#define EPSz 1e-6f

typedef __attribute__((ext_vector_type(16))) __bf16 v16bf;
typedef __attribute__((ext_vector_type(8)))  float  v8f;

union Frag256 { uint4 q[2]; v16bf v; };

// ---------- bf16 split helpers (RNE, finite inputs) ----------
__device__ __forceinline__ unsigned short f2bf_rne(float f) {
    unsigned int u = __float_as_uint(f);
    unsigned int r = u + 0x7fffu + ((u >> 16) & 1u);
    return (unsigned short)(r >> 16);
}
__device__ __forceinline__ float bf2f(unsigned short s) {
    return __uint_as_float(((unsigned int)s) << 16);
}

__device__ __forceinline__ float wave_sum32(float v) {
    #pragma unroll
    for (int off = 16; off > 0; off >>= 1) v += __shfl_xor(v, off, 32);
    return v;
}
__device__ __forceinline__ float rsum16(float v) {
    v += __shfl_xor(v, 1, 16);
    v += __shfl_xor(v, 2, 16);
    v += __shfl_xor(v, 4, 16);
    v += __shfl_xor(v, 8, 16);
    return v;
}
__device__ __forceinline__ float siluf(float x) {
    return x / (1.0f + expf(-x));
}

// ---------- f32 -> (bf16 hi, bf16 lo) decomposition ----------
__global__ __launch_bounds__(256) void decomp_kernel(const float* __restrict__ src,
        unsigned short* __restrict__ hi, unsigned short* __restrict__ lo, int nel) {
    int i = blockIdx.x * 256 + threadIdx.x;
    if (i >= nel) return;
    float v = src[i];
    unsigned short h = f2bf_rne(v);
    hi[i] = h;
    lo[i] = f2bf_rne(v - bf2f(h));
}

// ---------- bf16x3 WMMA GEMM:  C[M,N] = A[M,K] * W[N,K]^T  ----------
// Block = 128 threads (4 waves); block tile 64x64; wave tile 16x64 (4 N-subtiles).
// 12 WMMAs per K-block issued round-robin across the 4 independent accumulator
// chains so dependent WMMAs (same acc) are >=4 apart -> no v_nop hazard padding.
// out_mode 0: row-major [M,N].  out_mode 1: scatter to (b,h,s,hd) with row=b*S+s, col=h*64+hd.
__global__ __launch_bounds__(128) void gemm_bf16x3_kernel(
        const unsigned short* __restrict__ Ahi, const unsigned short* __restrict__ Alo,
        const unsigned short* __restrict__ Whi, const unsigned short* __restrict__ Wlo,
        float* __restrict__ Cout, int Kdim, int Ncols, int out_mode) {
    const int wv   = threadIdx.x >> 5;
    const int lane = threadIdx.x & 31;
    const int g    = lane >> 4;      // half-wave group
    const int n16  = lane & 15;
    const int rowBase = blockIdx.x * 64 + wv * 16;
    const int N0      = blockIdx.y * 64;
    const size_t aOff = (size_t)(rowBase + n16) * Kdim;

    v8f acc[4] = {};
    for (int kb = 0; kb < Kdim; kb += 32) {
        // A 16x32 bf16 fragment: lane(M=n16,g): K = kb+16g .. kb+16g+15 contiguous
        Frag256 ah, al;
        ah.q[0] = *(const uint4*)(Ahi + aOff + kb + 16 * g);
        ah.q[1] = *(const uint4*)(Ahi + aOff + kb + 16 * g + 8);
        al.q[0] = *(const uint4*)(Alo + aOff + kb + 16 * g);
        al.q[1] = *(const uint4*)(Alo + aOff + kb + 16 * g + 8);
        // B 32x16 fragments for the 4 N-subtiles:
        // lane(N=n16,g): W[N][kb+16g .. kb+16g+15] contiguous
        Frag256 bh[4], bl[4];
        #pragma unroll
        for (int s2 = 0; s2 < 4; ++s2) {
            size_t bOff = (size_t)(N0 + s2 * 16 + n16) * Kdim + kb + 16 * g;
            bh[s2].q[0] = *(const uint4*)(Whi + bOff);
            bh[s2].q[1] = *(const uint4*)(Whi + bOff + 8);
            bl[s2].q[0] = *(const uint4*)(Wlo + bOff);
            bl[s2].q[1] = *(const uint4*)(Wlo + bOff + 8);
        }
        // hi*hi, then hi*lo, then lo*hi -- interleaved across accumulators
        #pragma unroll
        for (int s2 = 0; s2 < 4; ++s2)
            acc[s2] = __builtin_amdgcn_wmma_f32_16x16x32_bf16(false, ah.v, false, bh[s2].v,
                                                              (short)0, acc[s2], false, false);
        #pragma unroll
        for (int s2 = 0; s2 < 4; ++s2)
            acc[s2] = __builtin_amdgcn_wmma_f32_16x16x32_bf16(false, ah.v, false, bl[s2].v,
                                                              (short)0, acc[s2], false, false);
        #pragma unroll
        for (int s2 = 0; s2 < 4; ++s2)
            acc[s2] = __builtin_amdgcn_wmma_f32_16x16x32_bf16(false, al.v, false, bh[s2].v,
                                                              (short)0, acc[s2], false, false);
    }
    #pragma unroll
    for (int s2 = 0; s2 < 4; ++s2) {
        int col = N0 + s2 * 16 + n16;
        #pragma unroll
        for (int r = 0; r < 8; ++r) {
            int row = rowBase + r + 8 * g;   // C vgpr r -> M = r + 8g
            float v = acc[s2][r];
            if (out_mode == 0) {
                Cout[(size_t)row * Ncols + col] = v;
            } else {
                int bb = row >> 11, st = row & (Sz - 1);
                int hh = col >> 6,  hd = col & 63;
                Cout[(((size_t)(bb * Hz + hh)) * Sz + st) * HDz + hd] = v;
            }
        }
    }
}

// ---------- RoPE in place on (b,h,s,hd) tensors; blockIdx.y selects Q/K/V ----------
__global__ __launch_bounds__(256) void rope_kernel(float* __restrict__ Q, float* __restrict__ K,
        float* __restrict__ V, const float* __restrict__ pf) {
    float* t = (blockIdx.y == 0) ? Q : (blockIdx.y == 1) ? K : V;
    int idx = blockIdx.x * 256 + threadIdx.x;            // pair index over B*H*S*32
    if (idx >= Bz * Hz * Sz * (HDz / 2)) return;
    int i  = idx & 31;
    int s  = (idx >> 5) & (Sz - 1);
    int bh = idx >> 16;
    float ang = pf[s * 32 + i];
    float c = cosf(ang), sn = sinf(ang);
    size_t base = ((size_t)bh * Sz + s) * HDz + 2 * i;
    float t1 = t[base], t2 = t[base + 1];
    t[base]     = t1 * c - t2 * sn;
    t[base + 1] = t1 * sn + t2 * c;
}

// ---------- eta = sigmoid(x . lr_W + lr_b)/HD * clip(1/(k+1)+g_scale, 0) ----------
__global__ __launch_bounds__(256) void eta_kernel(const float* __restrict__ x,
        const float* __restrict__ lrW, const float* __restrict__ lrb,
        const float* __restrict__ gsc, float* __restrict__ eta) {
    int gw   = blockIdx.x * 8 + (threadIdx.x >> 5);      // one wave per (b,h,n,k)
    int lane = threadIdx.x & 31;
    int k = gw & 15, n = (gw >> 4) & 127, h = (gw >> 11) & 15, b = gw >> 15;
    const float* xr = x + ((size_t)b * Sz + n * MBz + k) * DIMz;
    const float* wr = lrW + (size_t)h * DIMz;
    float acc = 0.f;
    for (int i = lane; i < DIMz; i += 32) acc += xr[i] * wr[i];
    acc = wave_sum32(acc);
    if (lane == 0) {
        float lr = 1.0f / (1.0f + expf(-(acc + lrb[h]))) * (1.0f / (float)HDz);
        float gs = 1.0f / (float)(k + 1) + gsc[k];
        if (gs < 0.f) gs = 0.f;
        eta[gw] = lr * gs;   // layout (b,h,n,k)
    }
}

// ---------- degree-3 B-spline basis over 15 knots -> 11 values ----------
__device__ __forceinline__ void bspline11(float x, const float* __restrict__ kn,
        const float (*rL)[13], const float (*rR)[13], float* __restrict__ outb) {
    float bb[14];
    #pragma unroll
    for (int i = 0; i < 14; ++i)
        bb[i] = (x >= kn[i] && x < kn[i + 1]) ? 1.0f : 0.0f;
    #pragma unroll
    for (int k = 1; k <= 3; ++k) {
        #pragma unroll
        for (int i = 0; i < 14 - k; ++i) {
            float left  = (x - kn[i]) * rL[k - 1][i];
            float right = (kn[i + k + 1] - x) * rR[k - 1][i];
            bb[i] = left * bb[i] + right * bb[i + 1];
        }
    }
    #pragma unroll
    for (int i = 0; i < NBz; ++i) outb[i] = bb[i];
}

// ---------- sequential TTT/KAN scan: one block per (b,h), 128 steps ----------
__global__ __launch_bounds__(256) void scan_kernel(
        const float* __restrict__ Q, const float* __restrict__ K, const float* __restrict__ V,
        const float* __restrict__ eta, const float* __restrict__ coeff,
        const float* __restrict__ tg, const float* __restrict__ tb,
        const float* __restrict__ knots, float* __restrict__ ypre) {
    __shared__ float sW[NBz][HDz];
    __shared__ float sBasK[MBz][HDz][NBz];
    __shared__ float sGz[MBz][HDz];
    __shared__ float sEi[MBz];
    __shared__ float sKn[NKz];
    __shared__ float sRcpL[3][13];
    __shared__ float sRcpR[3][13];

    const int t  = threadIdx.x;
    const int bh = blockIdx.x;
    const int b  = bh >> 4;
    const int h  = bh & 15;
    const int m  = t >> 4;             // token in minibatch
    const int d0 = (t & 15) * 4;       // 4 dims per thread

    for (int e = t; e < NBz * HDz; e += 256)
        sW[e / HDz][e % HDz] = coeff[(size_t)h * NBz * HDz + e];
    if (t < NKz) sKn[t] = knots[t];
    __syncthreads();
    if (t == 0) {
        for (int k = 1; k <= 3; ++k)
            for (int i = 0; i < 14 - k; ++i) {
                sRcpL[k - 1][i] = 1.0f / (sKn[i + k]     - sKn[i]);
                sRcpR[k - 1][i] = 1.0f / (sKn[i + k + 1] - sKn[i + 1]);
            }
    }
    float ga[4], be[4];
    #pragma unroll
    for (int j = 0; j < 4; ++j) {
        ga[j] = tg[h * HDz + d0 + j];
        be[j] = tb[h * HDz + d0 + j];
    }
    __syncthreads();

    const size_t headOff = (size_t)bh * Sz * HDz;
    for (int n = 0; n < NMBz; ++n) {
        size_t base = headOff + (size_t)(n * MBz + m) * HDz + d0;
        float4 k4 = *(const float4*)(K + base);
        float4 v4 = *(const float4*)(V + base);
        float4 q4 = *(const float4*)(Q + base);
        float ki[4] = {k4.x, k4.y, k4.z, k4.w};
        float vi[4] = {v4.x, v4.y, v4.z, v4.w};
        float qi[4] = {q4.x, q4.y, q4.z, q4.w};
        if (t < MBz) sEi[t] = eta[((size_t)bh * NMBz + n) * MBz + t];

        // ---- phase 1: basK, Zk = silu(K)+basK.W, layernorm-l2 backward -> gz ----
        float bk[4][NBz];
        #pragma unroll
        for (int j = 0; j < 4; ++j) {
            bspline11(ki[j], sKn, sRcpL, sRcpR, bk[j]);
            #pragma unroll
            for (int p = 0; p < NBz; ++p) sBasK[m][d0 + j][p] = bk[j][p];
        }
        float zk[4];
        #pragma unroll
        for (int j = 0; j < 4; ++j) {
            float s = 0.f;
            #pragma unroll
            for (int p = 0; p < NBz; ++p) s += bk[j][p] * sW[p][d0 + j];
            zk[j] = siluf(ki[j]) + s;
        }
        float loc = zk[0] + zk[1] + zk[2] + zk[3];
        float mu  = rsum16(loc) * (1.0f / HDz);
        loc = 0.f;
        #pragma unroll
        for (int j = 0; j < 4; ++j) { float dd = zk[j] - mu; loc += dd * dd; }
        float var = rsum16(loc) * (1.0f / HDz);
        float inv = 1.0f / sqrtf(var + EPSz);
        float xh[4], gh[4];
        loc = 0.f; float loc2 = 0.f;
        #pragma unroll
        for (int j = 0; j < 4; ++j) {
            xh[j] = (zk[j] - mu) * inv;
            gh[j] = (ga[j] * xh[j] + be[j] - (vi[j] - ki[j])) * ga[j];
            loc += gh[j]; loc2 += gh[j] * xh[j];
        }
        float mg  = rsum16(loc)  * (1.0f / HDz);
        float mgx = rsum16(loc2) * (1.0f / HDz);
        #pragma unroll
        for (int j = 0; j < 4; ++j)
            sGz[m][d0 + j] = (gh[j] - mg - xh[j] * mgx) * inv;
        __syncthreads();

        // ---- phase 2: Zq = silu(Q)+basQ.W - sum_{m'<=m} eta gz <basQ,basK>; y = Q + LN(Zq) ----
        float bq[4][NBz];
        float zq[4];
        #pragma unroll
        for (int j = 0; j < 4; ++j) {
            bspline11(qi[j], sKn, sRcpL, sRcpR, bq[j]);
            float s = 0.f;
            #pragma unroll
            for (int p = 0; p < NBz; ++p) s += bq[j][p] * sW[p][d0 + j];
            zq[j] = siluf(qi[j]) + s;
        }
        for (int m2 = 0; m2 <= m; ++m2) {
            float e = sEi[m2];
            #pragma unroll
            for (int j = 0; j < 4; ++j) {
                float s = 0.f;
                #pragma unroll
                for (int p = 0; p < NBz; ++p) s += bq[j][p] * sBasK[m2][d0 + j][p];
                zq[j] -= e * sGz[m2][d0 + j] * s;
            }
        }
        loc = zq[0] + zq[1] + zq[2] + zq[3];
        float mu2 = rsum16(loc) * (1.0f / HDz);
        loc = 0.f;
        #pragma unroll
        for (int j = 0; j < 4; ++j) { float dd = zq[j] - mu2; loc += dd * dd; }
        float var2 = rsum16(loc) * (1.0f / HDz);
        float inv2 = 1.0f / sqrtf(var2 + EPSz);
        float4 y4;
        y4.x = qi[0] + ga[0] * ((zq[0] - mu2) * inv2) + be[0];
        y4.y = qi[1] + ga[1] * ((zq[1] - mu2) * inv2) + be[1];
        y4.z = qi[2] + ga[2] * ((zq[2] - mu2) * inv2) + be[2];
        y4.w = qi[3] + ga[3] * ((zq[3] - mu2) * inv2) + be[3];
        *(float4*)(ypre + ((size_t)b * Sz + n * MBz + m) * DIMz + h * HDz + d0) = y4;
        __syncthreads();

        // ---- phase 3: W -= sum_{m'} eta[m'] gz[m',d] basK[m',d,p] ----
        for (int e2 = t; e2 < NBz * HDz; e2 += 256) {
            int p = e2 / HDz, d = e2 % HDz;
            float acc = 0.f;
            #pragma unroll
            for (int m2 = 0; m2 < MBz; ++m2)
                acc += sEi[m2] * sGz[m2][d] * sBasK[m2][d][p];
            sW[p][d] -= acc;
        }
        __syncthreads();
    }
}

// ---------- final layernorm over DIM + bf16 hi/lo decomposition ----------
__global__ __launch_bounds__(256) void finalln_kernel(const float* __restrict__ ypre,
        const float* __restrict__ pw, const float* __restrict__ pb,
        unsigned short* __restrict__ zhi, unsigned short* __restrict__ zlo) {
    __shared__ float red[8];
    const int row = blockIdx.x;          // b*S + s
    const int t   = threadIdx.x;
    const float* yr = ypre + (size_t)row * DIMz;
    float4 v4 = *(const float4*)(yr + t * 4);
    float v[4] = {v4.x, v4.y, v4.z, v4.w};

    float loc = v[0] + v[1] + v[2] + v[3];
    loc = wave_sum32(loc);
    if ((t & 31) == 0) red[t >> 5] = loc;
    __syncthreads();
    float tot = 0.f;
    #pragma unroll
    for (int i = 0; i < 8; ++i) tot += red[i];
    float mu = tot * (1.0f / DIMz);
    __syncthreads();

    loc = 0.f;
    #pragma unroll
    for (int j = 0; j < 4; ++j) { float dd = v[j] - mu; loc += dd * dd; }
    loc = wave_sum32(loc);
    if ((t & 31) == 0) red[t >> 5] = loc;
    __syncthreads();
    tot = 0.f;
    #pragma unroll
    for (int i = 0; i < 8; ++i) tot += red[i];
    float inv = 1.0f / sqrtf(tot * (1.0f / DIMz) + EPSz);

    #pragma unroll
    for (int j = 0; j < 4; ++j) {
        int d = t * 4 + j;
        float z = (v[j] - mu) * inv * pw[d] + pb[d];
        unsigned short hh = f2bf_rne(z);
        zhi[(size_t)row * DIMz + d] = hh;
        zlo[(size_t)row * DIMz + d] = f2bf_rne(z - bf2f(hh));
    }
}

// =======================================================================
extern "C" void kernel_launch(void* const* d_in, const int* in_sizes, int n_in,
                              void* d_out, int out_size, void* d_ws, size_t ws_size,
                              hipStream_t stream) {
    const float* x     = (const float*)d_in[0];
    const float* pf    = (const float*)d_in[1];
    const float* Wq    = (const float*)d_in[2];
    const float* Wk    = (const float*)d_in[3];
    const float* Wv    = (const float*)d_in[4];
    const float* Wo    = (const float*)d_in[5];
    const float* lrW   = (const float*)d_in[6];
    const float* lrb   = (const float*)d_in[7];
    const float* gsc   = (const float*)d_in[8];
    const float* tg    = (const float*)d_in[9];
    const float* tb    = (const float*)d_in[10];
    const float* pw    = (const float*)d_in[11];
    const float* pb    = (const float*)d_in[12];
    const float* coeff = (const float*)d_in[13];
    const float* knots = (const float*)d_in[14];
    float* out = (float*)d_out;

    const size_t NX = (size_t)Bz * Sz * DIMz;   // 8,388,608
    const size_t NW = (size_t)DIMz * DIMz;      // 1,048,576

    char* w = (char*)d_ws;
    size_t off = 0;
    auto take = [&](size_t bytes) { size_t r = off; off += (bytes + 255) & ~(size_t)255; return r; };
    unsigned short* xhi  = (unsigned short*)(w + take(NX * 2));
    unsigned short* xlo  = (unsigned short*)(w + take(NX * 2));
    unsigned short* wqhi = (unsigned short*)(w + take(NW * 2));
    unsigned short* wqlo = (unsigned short*)(w + take(NW * 2));
    unsigned short* wkhi = (unsigned short*)(w + take(NW * 2));
    unsigned short* wklo = (unsigned short*)(w + take(NW * 2));
    unsigned short* wvhi = (unsigned short*)(w + take(NW * 2));
    unsigned short* wvlo = (unsigned short*)(w + take(NW * 2));
    unsigned short* wohi = (unsigned short*)(w + take(NW * 2));
    unsigned short* wolo = (unsigned short*)(w + take(NW * 2));
    float* QR   = (float*)(w + take(NX * 4));
    float* KR   = (float*)(w + take(NX * 4));
    float* VR   = (float*)(w + take(NX * 4));
    float* ETA  = (float*)(w + take((size_t)Bz * Hz * NMBz * MBz * 4));
    float* YPRE = (float*)(w + take(NX * 4));
    // final-LN output reuses x's bf16 buffers (x_hi/x_lo dead after QKV GEMMs)
    unsigned short* zhi = xhi;
    unsigned short* zlo = xlo;

    // 1) decompose x and weights into bf16 hi/lo
    decomp_kernel<<<(int)(NX / 256), 256, 0, stream>>>(x,  xhi,  xlo,  (int)NX);
    decomp_kernel<<<(int)(NW / 256), 256, 0, stream>>>(Wq, wqhi, wqlo, (int)NW);
    decomp_kernel<<<(int)(NW / 256), 256, 0, stream>>>(Wk, wkhi, wklo, (int)NW);
    decomp_kernel<<<(int)(NW / 256), 256, 0, stream>>>(Wv, wvhi, wvlo, (int)NW);
    decomp_kernel<<<(int)(NW / 256), 256, 0, stream>>>(Wo, wohi, wolo, (int)NW);

    // 2) QKV projections via bf16x3 WMMA, scattered to (b,h,s,hd)
    dim3 ggrid((Bz * Sz) / 64, DIMz / 64);
    gemm_bf16x3_kernel<<<ggrid, 128, 0, stream>>>(xhi, xlo, wqhi, wqlo, QR, DIMz, DIMz, 1);
    gemm_bf16x3_kernel<<<ggrid, 128, 0, stream>>>(xhi, xlo, wkhi, wklo, KR, DIMz, DIMz, 1);
    gemm_bf16x3_kernel<<<ggrid, 128, 0, stream>>>(xhi, xlo, wvhi, wvlo, VR, DIMz, DIMz, 1);

    // 3) RoPE (in place; buffers fully rewritten by GEMMs each call)
    dim3 rgrid((Bz * Hz * Sz * (HDz / 2)) / 256, 3);
    rope_kernel<<<rgrid, 256, 0, stream>>>(QR, KR, VR, pf);

    // 4) eta gate (one wave per (b,h,n,k) dot product)
    eta_kernel<<<(Bz * Hz * NMBz * MBz) / 8, 256, 0, stream>>>(x, lrW, lrb, gsc, ETA);

    // 5) sequential scan: one workgroup per (b,h)
    scan_kernel<<<Bz * Hz, 256, 0, stream>>>(QR, KR, VR, ETA, coeff, tg, tb, knots, YPRE);

    // 6) final layernorm + bf16 decomposition
    finalln_kernel<<<Bz * Sz, 256, 0, stream>>>(YPRE, pw, pb, zhi, zlo);

    // 7) output projection via bf16x3 WMMA
    gemm_bf16x3_kernel<<<ggrid, 128, 0, stream>>>(zhi, zlo, wohi, wolo, out, DIMz, DIMz, 0);

    (void)in_sizes; (void)n_in; (void)out_size; (void)ws_size;
}
